// CodeSageAttention_41609643164075
// MI455X (gfx1250) — compile-verified, matrix operand order checked by reading, not run on hardware
//
#include <hip/hip_runtime.h>
#include <hip/hip_bf16.h>

typedef __attribute__((ext_vector_type(16))) __bf16 v16bf;
typedef __attribute__((ext_vector_type(8)))  __bf16 v8bf;
typedef __attribute__((ext_vector_type(4)))  __bf16 v4bf;
typedef __attribute__((ext_vector_type(8)))  float  v8f;

typedef unsigned int u32x4 __attribute__((ext_vector_type(4)));
typedef int          i32x4 __attribute__((ext_vector_type(4)));
typedef int          i32x8 __attribute__((ext_vector_type(8)));

#define BATCH 2
#define SEQ   2048
#define HID   1024
#define NHEAD 16
#define HDIM  64

// ---------------------------------------------------------------------------
// TDM: copy a 2D bf16 tile (tile_d0 elems per row, tile_d1 rows, row stride
// stride_elems) from global memory to LDS (rows packed back-to-back).
// Descriptor per CDNA5 ISA ch.8.  Completion tracked via TENSORcnt.
// ---------------------------------------------------------------------------
__device__ __forceinline__
void tdm_load_bf16(const void* gsrc, unsigned lds_off,
                   unsigned tile_d0, unsigned tile_d1, unsigned stride_elems)
{
    unsigned long long ga = (unsigned long long)(size_t)gsrc;
    u32x4 g0;
    g0[0] = 1u;                                                // count=1, user desc
    g0[1] = lds_off;                                           // LDS byte address
    g0[2] = (unsigned)(ga & 0xFFFFFFFFu);                      // global_addr[31:0]
    g0[3] = (unsigned)((ga >> 32) & 0x01FFFFFFu) | (2u << 30); // addr[56:32] | type=2

    const unsigned td0 = 1u << 30;     // generous tensor bounds: no OOB clipping
    const unsigned td1 = 1u << 20;
    i32x8 g1;
    g1[0] = (1 << 16);                                              // data_size=2B
    g1[1] = (int)((td0 & 0xFFFFu) << 16);                           // tensor_dim0 lo
    g1[2] = (int)(((td0 >> 16) & 0xFFFFu) | ((td1 & 0xFFFFu) << 16));
    g1[3] = (int)(((td1 >> 16) & 0xFFFFu) | (tile_d0 << 16));       // tile_dim0
    g1[4] = (int)(tile_d1 & 0xFFFFu);                               // tile_dim1
    g1[5] = (int)stride_elems;                                      // dim0 stride lo
    g1[6] = 0;
    g1[7] = 0;

    i32x4 z4 = {0, 0, 0, 0};
#if defined(__clang_major__) && (__clang_major__ >= 23)
    i32x8 z8 = {0, 0, 0, 0, 0, 0, 0, 0};
    __builtin_amdgcn_tensor_load_to_lds(g0, g1, z4, z4, z8, 0);
#else
    __builtin_amdgcn_tensor_load_to_lds(g0, g1, z4, z4, 0);
#endif
}

// ---------------------------------------------------------------------------
// Kernel 1: fused QKV projection.  X[4096,1024] @ W[1024,3072] + b
// ---------------------------------------------------------------------------
__global__ __launch_bounds__(256)
void qkv_kernel(const float* __restrict__ X, const float* __restrict__ W,
                const float* __restrict__ bias,
                __bf16* __restrict__ qb, __bf16* __restrict__ kb,
                __bf16* __restrict__ vb)
{
    __shared__ __bf16 As[128 * 64];   // [row][k]
    __shared__ __bf16 Bs[64 * 64];    // [n][k] transposed

    const int tid   = threadIdx.x;
    const int lane  = tid & 31;
    const int wave  = tid >> 5;
    const int group = lane >> 4;
    const int lcol  = lane & 15;
    const int n0    = blockIdx.x * 64;
    const int m0    = blockIdx.y * 128;

    // 4x4 micro-tile coords for transposed B staging
    const int bk4 = (tid >> 4) << 2;     // k  in [0,64) step 4
    const int bn4 = (tid & 15) << 2;     // n  in [0,64) step 4

    v8f acc[4] = {};

    for (int k0 = 0; k0 < HID; k0 += 64) {
        if (k0 + 64 < HID) {
            const int pr = tid >> 4, pc = (tid & 15) << 2;
            __builtin_prefetch(X + (size_t)(m0 + pr) * HID + k0 + 64 + pc, 0, 3);
            __builtin_prefetch(W + (size_t)(k0 + 64 + pr) * (3 * HID) + n0 + pc, 0, 3);
        }
        // ---- stage A tile (128x64 f32 -> bf16), packed b64 stores ----
        #pragma unroll
        for (int i = 0; i < 8; ++i) {
            int c   = tid + i * 256;
            int row = c >> 4;
            int kc  = (c & 15) << 2;
            const float4 f = *(const float4*)(X + (size_t)(m0 + row) * HID + k0 + kc);
            v4bf p = { (__bf16)f.x, (__bf16)f.y, (__bf16)f.z, (__bf16)f.w };
            *(v4bf*)&As[row * 64 + kc] = p;
        }
        // ---- stage B tile transposed, 4x4 micro-tile, packed b64 stores ----
        {
            float cv[4][4];
            #pragma unroll
            for (int r = 0; r < 4; ++r) {
                const float4 f =
                    *(const float4*)(W + (size_t)(k0 + bk4 + r) * (3 * HID) + n0 + bn4);
                cv[r][0] = f.x; cv[r][1] = f.y; cv[r][2] = f.z; cv[r][3] = f.w;
            }
            #pragma unroll
            for (int c = 0; c < 4; ++c) {
                v4bf p = { (__bf16)cv[0][c], (__bf16)cv[1][c],
                           (__bf16)cv[2][c], (__bf16)cv[3][c] };
                *(v4bf*)&Bs[(bn4 + c) * 64 + bk4] = p;
            }
        }
        __syncthreads();

        const int arow = wave * 16 + lcol;
        #pragma unroll
        for (int ks = 0; ks < 64; ks += 32) {
            v8bf alo = *(const v8bf*)&As[arow * 64 + ks + group * 8];
            v8bf ahi = *(const v8bf*)&As[arow * 64 + ks + 16 + group * 8];
            v16bf afrag;
            #pragma unroll
            for (int e = 0; e < 8; ++e) { afrag[e] = alo[e]; afrag[8 + e] = ahi[e]; }

            #pragma unroll
            for (int t = 0; t < 4; ++t) {
                const int bn = t * 16 + lcol;
                v8bf blo = *(const v8bf*)&Bs[bn * 64 + ks + group * 8];
                v8bf bhi = *(const v8bf*)&Bs[bn * 64 + ks + 16 + group * 8];
                v16bf bfrag;
                #pragma unroll
                for (int e = 0; e < 8; ++e) { bfrag[e] = blo[e]; bfrag[8 + e] = bhi[e]; }
                acc[t] = __builtin_amdgcn_wmma_f32_16x16x32_bf16(
                    false, afrag, false, bfrag, (short)0, acc[t], false, false);
            }
        }
        __syncthreads();
    }

    // ---- epilogue: +bias, scatter to head layout, fold softmax scale into q ----
    const int mrow0 = m0 + wave * 16 + group * 8;
    #pragma unroll
    for (int t = 0; t < 4; ++t) {
        const int n      = n0 + t * 16 + lcol;
        const float bv   = bias[n];
        const int region = n >> 10;
        const int rem    = n & 1023;
        const int h      = rem >> 6;
        const int d      = rem & 63;
        __bf16* dstbase  = (region == 0) ? qb : (region == 1 ? kb : vb);
        const float scale = (region == 0) ? 0.125f : 1.0f;
        #pragma unroll
        for (int j = 0; j < 8; ++j) {
            const int m = mrow0 + j;
            const int b = m >> 11;
            const int s = m & 2047;
            dstbase[(((size_t)(b * NHEAD + h)) * SEQ + s) * HDIM + d] =
                (__bf16)((acc[t][j] + bv) * scale);
        }
    }
}

// ---------------------------------------------------------------------------
// Kernel 2: flash attention per (b,h), 64-key blocks.  K tile via TDM DMA,
// V staged transposed with packed b64 stores, all matmuls via WMMA.
// ---------------------------------------------------------------------------
__global__ __launch_bounds__(128)
void attn_kernel(const __bf16* __restrict__ Q, const __bf16* __restrict__ K,
                 const __bf16* __restrict__ V, const float* __restrict__ mask,
                 __bf16* __restrict__ ctx)
{
    __shared__ __bf16 Ks[64 * 64];        // [key_local][d]  (TDM destination)
    __shared__ __bf16 Vs[64 * 64];        // [d][key_local]  (transposed)
    __shared__ __bf16 Ps[4][16 * 64];     // per-wave P staging [row][key_local]

    const int tid   = threadIdx.x;
    const int lane  = tid & 31;
    const int wave  = tid >> 5;
    const int group = lane >> 4;
    const int lcol  = lane & 15;
    const int bh    = blockIdx.y;
    const int b     = bh >> 4;
    const int q0    = blockIdx.x * 64 + wave * 16;

    const __bf16* Qh = Q + (size_t)bh * SEQ * HDIM;
    const __bf16* Kh = K + (size_t)bh * SEQ * HDIM;
    const __bf16* Vh = V + (size_t)bh * SEQ * HDIM;
    const float*  Mh = mask + (size_t)b * SEQ;

    const unsigned ks_lds = (unsigned)(size_t)(void*)&Ks[0];

    // 4x4 micro-tile coords for transposed V staging
    const int vd4 = (tid & 15) << 2;     // d   in [0,64) step 4
    const int vk4 = (tid >> 4) << 2;     // key in [0,32) step 4 (x2 passes)

    // Q fragments, resident for the whole key loop.
    v16bf aq[2];
    {
        const int qrow = q0 + lcol;
        #pragma unroll
        for (int i = 0; i < 2; ++i) {
            v8bf lo = *(const v8bf*)(Qh + (size_t)qrow * HDIM + i * 32 + group * 8);
            v8bf hi = *(const v8bf*)(Qh + (size_t)qrow * HDIM + i * 32 + 16 + group * 8);
            #pragma unroll
            for (int e = 0; e < 8; ++e) { aq[i][e] = lo[e]; aq[i][8 + e] = hi[e]; }
        }
    }

    v8f acc[4] = {};
    v8f mmax, lsum;
    #pragma unroll
    for (int j = 0; j < 8; ++j) { mmax[j] = -1e30f; lsum[j] = 0.0f; }

    for (int kb0 = 0; kb0 < SEQ; kb0 += 64) {
        __syncthreads();   // all readers of Ks/Vs from previous iter are done

        // ---- K tile (64x64, contiguous 8KB): TDM async DMA ----
        if (wave == 0)
            tdm_load_bf16(Kh + (size_t)kb0 * HDIM, ks_lds, 64 * HDIM, 1, 64 * HDIM);

        // ---- V tile: transposed staging Vs[d][key], packed b64 stores ----
        #pragma unroll
        for (int pass = 0; pass < 2; ++pass) {
            const int key = vk4 + pass * 32;
            v4bf rr[4];
            #pragma unroll
            for (int r = 0; r < 4; ++r)
                rr[r] = *(const v4bf*)(Vh + (size_t)(kb0 + key + r) * HDIM + vd4);
            #pragma unroll
            for (int c = 0; c < 4; ++c) {
                v4bf p = { rr[0][c], rr[1][c], rr[2][c], rr[3][c] };
                *(v4bf*)&Vs[(vd4 + c) * 64 + key] = p;
            }
            if (kb0 + 64 < SEQ)
                __builtin_prefetch(Vh + (size_t)(kb0 + 64 + key) * HDIM + vd4, 0, 3);
        }

        if (wave == 0)
            __builtin_amdgcn_s_wait_tensorcnt((short)0);   // TDM done
        __syncthreads();   // Ks (TDM) and Vs visible to all waves

        // ---- scores: four 16x16 key tiles from LDS ----
        v8f s[4];
        #pragma unroll
        for (int kt = 0; kt < 4; ++kt) {
            const int keyloc = kt * 16 + lcol;
            v8f sc = {};
            #pragma unroll
            for (int i = 0; i < 2; ++i) {
                v8bf lo = *(const v8bf*)&Ks[keyloc * HDIM + i * 32 + group * 8];
                v8bf hi = *(const v8bf*)&Ks[keyloc * HDIM + i * 32 + 16 + group * 8];
                v16bf bk;
                #pragma unroll
                for (int e = 0; e < 8; ++e) { bk[e] = lo[e]; bk[8 + e] = hi[e]; }
                sc = __builtin_amdgcn_wmma_f32_16x16x32_bf16(
                    false, aq[i], false, bk, (short)0, sc, false, false);
            }
            const float mv = Mh[kb0 + keyloc];
            #pragma unroll
            for (int j = 0; j < 8; ++j) sc[j] += mv;
            s[kt] = sc;
        }

        // ---- online softmax ----
        v8f mnew;
        #pragma unroll
        for (int j = 0; j < 8; ++j) {
            float t = fmaxf(fmaxf(s[0][j], s[1][j]), fmaxf(s[2][j], s[3][j]));
            #pragma unroll
            for (int m = 1; m < 16; m <<= 1) t = fmaxf(t, __shfl_xor(t, m, 32));
            mnew[j] = fmaxf(mmax[j], t);
        }
        #pragma unroll
        for (int j = 0; j < 8; ++j) {
            const float alpha = __expf(mmax[j] - mnew[j]);
            lsum[j] *= alpha;
            #pragma unroll
            for (int t2 = 0; t2 < 4; ++t2) acc[t2][j] *= alpha;
            mmax[j] = mnew[j];
            float ps = 0.0f;
            #pragma unroll
            for (int kt = 0; kt < 4; ++kt) {
                s[kt][j] = __expf(s[kt][j] - mnew[j]);
                ps += s[kt][j];
            }
            lsum[j] += ps;                 // per-lane partial; reduced at end
        }

        // ---- C-layout -> A-layout for P via per-wave LDS bounce ----
        __bf16* P = Ps[wave];
        #pragma unroll
        for (int j = 0; j < 8; ++j) {
            const int prow = group * 8 + j;
            #pragma unroll
            for (int kt = 0; kt < 4; ++kt)
                P[prow * 64 + kt * 16 + lcol] = (__bf16)s[kt][j];
        }

        // ---- context accumulation: O += P @ V over two 32-key chunks ----
        #pragma unroll
        for (int kc = 0; kc < 2; ++kc) {
            v16bf ap;   // same-wave LDS RAW: DScnt keeps this ordered
            {
                v8bf lo = *(const v8bf*)&P[lcol * 64 + kc * 32 + group * 8];
                v8bf hi = *(const v8bf*)&P[lcol * 64 + kc * 32 + 16 + group * 8];
                #pragma unroll
                for (int e = 0; e < 8; ++e) { ap[e] = lo[e]; ap[8 + e] = hi[e]; }
            }
            #pragma unroll
            for (int t2 = 0; t2 < 4; ++t2) {
                const int dcol = t2 * 16 + lcol;
                v8bf lo = *(const v8bf*)&Vs[dcol * 64 + kc * 32 + group * 8];
                v8bf hi = *(const v8bf*)&Vs[dcol * 64 + kc * 32 + 16 + group * 8];
                v16bf bv;
                #pragma unroll
                for (int e = 0; e < 8; ++e) { bv[e] = lo[e]; bv[8 + e] = hi[e]; }
                acc[t2] = __builtin_amdgcn_wmma_f32_16x16x32_bf16(
                    false, ap, false, bv, (short)0, acc[t2], false, false);
            }
        }
    }

    // ---- finalize ----
    #pragma unroll
    for (int j = 0; j < 8; ++j) {
        float t = lsum[j];
        #pragma unroll
        for (int m = 1; m < 16; m <<= 1) t += __shfl_xor(t, m, 32);
        const float inv = 1.0f / t;
        #pragma unroll
        for (int t2 = 0; t2 < 4; ++t2) acc[t2][j] *= inv;
    }
    const int h = bh & 15;
    #pragma unroll
    for (int t2 = 0; t2 < 4; ++t2) {
        const int d = t2 * 16 + lcol;
        #pragma unroll
        for (int j = 0; j < 8; ++j) {
            const int srow = q0 + group * 8 + j;
            ctx[((size_t)(b * SEQ + srow)) * HID + h * HDIM + d] = (__bf16)acc[t2][j];
        }
    }
}

// ---------------------------------------------------------------------------
// Kernel 3: output projection.  ctx_bf16[4096,1024] @ W[1024,1024] + b -> f32
// A tile delivered by TDM (already bf16 in workspace), B staged transposed.
// ---------------------------------------------------------------------------
__global__ __launch_bounds__(256)
void proj_kernel(const __bf16* __restrict__ Xc, const float* __restrict__ W,
                 const float* __restrict__ bias, float* __restrict__ out)
{
    __shared__ __bf16 As[128 * 64];   // [row][k]  (TDM destination)
    __shared__ __bf16 Bs[64 * 64];    // [n][k] transposed

    const int tid   = threadIdx.x;
    const int lane  = tid & 31;
    const int wave  = tid >> 5;
    const int group = lane >> 4;
    const int lcol  = lane & 15;
    const int n0    = blockIdx.x * 64;
    const int m0    = blockIdx.y * 128;

    const int bk4 = (tid >> 4) << 2;
    const int bn4 = (tid & 15) << 2;

    const unsigned as_lds = (unsigned)(size_t)(void*)&As[0];

    v8f acc[4] = {};

    for (int k0 = 0; k0 < HID; k0 += 64) {
        // ---- A tile (128 rows x 64 k, bf16): TDM 2D async DMA ----
        if (wave == 0)
            tdm_load_bf16(Xc + (size_t)m0 * HID + k0, as_lds, 64, 128, HID);

        if (k0 + 64 < HID) {
            const int pr = tid >> 4, pc = (tid & 15) << 2;
            __builtin_prefetch(W + (size_t)(k0 + 64 + pr) * HID + n0 + pc, 0, 3);
        }
        // ---- stage B f32 -> bf16, transposed, packed b64 stores ----
        {
            float cv[4][4];
            #pragma unroll
            for (int r = 0; r < 4; ++r) {
                const float4 f =
                    *(const float4*)(W + (size_t)(k0 + bk4 + r) * HID + n0 + bn4);
                cv[r][0] = f.x; cv[r][1] = f.y; cv[r][2] = f.z; cv[r][3] = f.w;
            }
            #pragma unroll
            for (int c = 0; c < 4; ++c) {
                v4bf p = { (__bf16)cv[0][c], (__bf16)cv[1][c],
                           (__bf16)cv[2][c], (__bf16)cv[3][c] };
                *(v4bf*)&Bs[(bn4 + c) * 64 + bk4] = p;
            }
        }

        if (wave == 0)
            __builtin_amdgcn_s_wait_tensorcnt((short)0);   // TDM done
        __syncthreads();

        const int arow = wave * 16 + lcol;
        #pragma unroll
        for (int ks = 0; ks < 64; ks += 32) {
            v8bf alo = *(const v8bf*)&As[arow * 64 + ks + group * 8];
            v8bf ahi = *(const v8bf*)&As[arow * 64 + ks + 16 + group * 8];
            v16bf afrag;
            #pragma unroll
            for (int e = 0; e < 8; ++e) { afrag[e] = alo[e]; afrag[8 + e] = ahi[e]; }

            #pragma unroll
            for (int t = 0; t < 4; ++t) {
                const int bn = t * 16 + lcol;
                v8bf blo = *(const v8bf*)&Bs[bn * 64 + ks + group * 8];
                v8bf bhi = *(const v8bf*)&Bs[bn * 64 + ks + 16 + group * 8];
                v16bf bfrag;
                #pragma unroll
                for (int e = 0; e < 8; ++e) { bfrag[e] = blo[e]; bfrag[8 + e] = bhi[e]; }
                acc[t] = __builtin_amdgcn_wmma_f32_16x16x32_bf16(
                    false, afrag, false, bfrag, (short)0, acc[t], false, false);
            }
        }
        __syncthreads();
    }

    const int mrow0 = m0 + wave * 16 + group * 8;
    #pragma unroll
    for (int t = 0; t < 4; ++t) {
        const int n    = n0 + t * 16 + lcol;
        const float bv = bias[n];
        #pragma unroll
        for (int j = 0; j < 8; ++j) {
            const int m = mrow0 + j;
            out[(size_t)m * HID + n] = acc[t][j] + bv;
        }
    }
}

// ---------------------------------------------------------------------------
extern "C" void kernel_launch(void* const* d_in, const int* in_sizes, int n_in,
                              void* d_out, int out_size, void* d_ws, size_t ws_size,
                              hipStream_t stream)
{
    (void)in_sizes; (void)n_in; (void)out_size; (void)ws_size;

    const float* hidden = (const float*)d_in[0];
    const float* mask   = (const float*)d_in[1];
    const float* attw   = (const float*)d_in[2];
    const float* attb   = (const float*)d_in[3];
    const float* projw  = (const float*)d_in[4];
    const float* projb  = (const float*)d_in[5];

    const size_t head_elems = (size_t)BATCH * NHEAD * SEQ * HDIM;
    __bf16* qb  = (__bf16*)d_ws;
    __bf16* kb  = qb + head_elems;
    __bf16* vb  = kb + head_elems;
    __bf16* ctx = vb + head_elems;
    float*  out = (float*)d_out;

    dim3 g1(3 * HID / 64, BATCH * SEQ / 128);
    qkv_kernel<<<g1, 256, 0, stream>>>(hidden, attw, attb, qb, kb, vb);

    dim3 g2(SEQ / 64, BATCH * NHEAD);
    attn_kernel<<<g2, 128, 0, stream>>>(qb, kb, vb, mask, ctx);

    dim3 g3(HID / 64, BATCH * SEQ / 128);
    proj_kernel<<<g3, 256, 0, stream>>>(ctx, projw, projb, out);
}
